// SinkhornDivergence_10780367913245
// MI455X (gfx1250) — matrix operand order repaired and checked by reading, not compiled, not run on hardware
//
#include <hip/hip_runtime.h>

typedef __attribute__((ext_vector_type(16))) _Float16 v16h;
typedef __attribute__((ext_vector_type(8)))  _Float16 v8h;
typedef __attribute__((ext_vector_type(8)))  float    v8f;

#define NROWS 4096
#define KDIM  512
#define KPAD  520                 // LDS row stride in halves (bank-conflict padding)
#define EPS_F     0.1f
#define INV_EPS   10.0f
#define C_CLAMP_F 1.0e6f
#define LOG2E_F   1.4426950408889634f
#define LN2_F     0.6931471805599453f
#define SCL_F     (INV_EPS * LOG2E_F)   // 14.426950409

__device__ __forceinline__ float fast_exp2(float x) {
#if __has_builtin(__builtin_amdgcn_exp2f)
  return __builtin_amdgcn_exp2f(x);
#else
  return exp2f(x);
#endif
}
__device__ __forceinline__ float fast_log2(float x) {
#if __has_builtin(__builtin_amdgcn_logf)
  return __builtin_amdgcn_logf(x);
#else
  return log2f(x);
#endif
}
__device__ __forceinline__ void wait_async0() {
#if __has_builtin(__builtin_amdgcn_s_wait_asynccnt)
  __builtin_amdgcn_s_wait_asynccnt(0);
#else
  asm volatile("s_wait_asynccnt 0x0" ::: "memory");
#endif
}

// ------------------------------------------------------------------
// fp32 -> f16 conversion
// ------------------------------------------------------------------
__global__ void cvt_f16_kernel(const float* __restrict__ in,
                               _Float16* __restrict__ out, int n) {
  int i = blockIdx.x * blockDim.x + threadIdx.x;
  if (i < n) out[i] = (_Float16)in[i];
}

// ------------------------------------------------------------------
// row squared norms: one wave32 per row of (NROWS x KDIM)
// ------------------------------------------------------------------
__global__ void row_norms_kernel(const float* __restrict__ in,
                                 float* __restrict__ out) {
  int wave = (blockIdx.x * blockDim.x + threadIdx.x) >> 5;
  int lane = threadIdx.x & 31;
  const float* r = in + (size_t)wave * KDIM;
  float s = 0.0f;
  for (int k = lane * 4; k < KDIM; k += 128) {
    float4 v = *(const float4*)(r + k);
    s += v.x * v.x + v.y * v.y + v.z * v.z + v.w * v.w;
  }
  #pragma unroll
  for (int off = 16; off; off >>= 1) s += __shfl_xor(s, off, 32);
  if (lane == 0) out[wave] = s;
}

// ------------------------------------------------------------------
// Cost-matrix GEMM via WMMA f16:
//   C[i,j] = clamp(a2[i] + b2[j] - 2 * dot(A_i, B_j), 0, 1e6)
// Block = 8 waves -> 128 rows x 64 cols.
// B panel (64 rows x 512) staged once per block into LDS with
// async global->LDS loads (ASYNCcnt path), WMMA B-frags read via DS.
// ------------------------------------------------------------------
union AFrag { v16h v; v8h h[2]; };

__global__ void gemm_cost_kernel(const _Float16* __restrict__ A,
                                 const _Float16* __restrict__ B,
                                 const float* __restrict__ a2,
                                 const float* __restrict__ b2,
                                 float* __restrict__ C) {
  __shared__ __align__(32) _Float16 bpanel[64 * KPAD];   // ~65 KB

  const int tid  = threadIdx.x;
  const int wave = tid >> 5;
  const int lane = tid & 31;
  const int m16  = lane & 15;
  const int hi   = lane >> 4;
  const int M0   = (blockIdx.y * 8 + wave) * 16;
  const int N0   = blockIdx.x * 64;

  // ---- async-stage B panel: 64 rows x 512 halves (contiguous in B) ----
  {
    const _Float16* gpanel = B + (size_t)N0 * KDIM;
    // 4096 chunks of 16B; 256 threads -> 16 chunks each
    for (int c = tid; c < 64 * (KDIM / 8); c += 256) {
      int r = c >> 6;                 // row within panel
      int k = (c & 63) * 8;           // half offset within row
      const _Float16* gsrc = gpanel + (size_t)r * KDIM + k;
      unsigned int ldsoff = (unsigned int)(uintptr_t)(&bpanel[r * KPAD + k]);
      asm volatile("global_load_async_to_lds_b128 %0, %1, off"
                   :: "v"(ldsoff), "v"(gsrc) : "memory");
    }
    wait_async0();
    __syncthreads();
  }

  v8f acc[4] = {};
  const _Float16* arow = A + (size_t)(M0 + m16) * KDIM;

  for (int k0 = 0; k0 < KDIM; k0 += 32) {
    AFrag a;
    a.v = *(const v16h*)(arow + k0 + hi * 16);         // K = k0 + hi*16 .. +15
    v16h bf0 = *(const v16h*)(&bpanel[( 0 + m16) * KPAD + hi * 16 + k0]);
    v16h bf1 = *(const v16h*)(&bpanel[(16 + m16) * KPAD + hi * 16 + k0]);
    v16h bf2 = *(const v16h*)(&bpanel[(32 + m16) * KPAD + hi * 16 + k0]);
    v16h bf3 = *(const v16h*)(&bpanel[(48 + m16) * KPAD + hi * 16 + k0]);
    acc[0] = __builtin_amdgcn_wmma_f32_16x16x32_f16(false, a.v, false, bf0,
                                                    (short)0, acc[0], false, false);
    acc[1] = __builtin_amdgcn_wmma_f32_16x16x32_f16(false, a.v, false, bf1,
                                                    (short)0, acc[1], false, false);
    acc[2] = __builtin_amdgcn_wmma_f32_16x16x32_f16(false, a.v, false, bf2,
                                                    (short)0, acc[2], false, false);
    acc[3] = __builtin_amdgcn_wmma_f32_16x16x32_f16(false, a.v, false, bf3,
                                                    (short)0, acc[3], false, false);
  }

  float a2r[8];
  #pragma unroll
  for (int r = 0; r < 8; ++r) a2r[r] = a2[M0 + hi * 8 + r];

  #pragma unroll
  for (int t = 0; t < 4; ++t) {
    int col  = N0 + 16 * t + m16;
    float bn = b2[col];
    #pragma unroll
    for (int r = 0; r < 8; ++r) {
      float v = a2r[r] + bn - 2.0f * acc[t][r];
      v = fminf(fmaxf(v, 0.0f), C_CLAMP_F);
      C[(size_t)(M0 + hi * 8 + r) * NROWS + col] = v;
    }
  }
}

// ------------------------------------------------------------------
// Row-wise streaming logsumexp (one Sinkhorn half-step), log2 domain:
//   vout[r] = -EPS * ( logw + ln2 * log2sumexp2_k( (vin[k]-C[r,k])*10*log2e ) )
// One wave32 per row; dual vector pre-scaled in LDS; online chunked LSE.
// Exactly one v_exp_f32 per element.
// ------------------------------------------------------------------
__global__ void lse_rows_kernel(const float* __restrict__ C,
                                const float* __restrict__ vin,
                                float* __restrict__ vout, float logw) {
  __shared__ float sv[NROWS];
  for (int k = threadIdx.x; k < NROWS; k += blockDim.x)
    sv[k] = vin[k] * SCL_F;
  __syncthreads();

  const int wave = threadIdx.x >> 5;
  const int lane = threadIdx.x & 31;
  const int row  = blockIdx.x * 8 + wave;
  const float* crow = C + (size_t)row * NROWS;

  float m = -1.0e30f, s = 0.0f;
  for (int t = 0; t < NROWS; t += 512) {
    int base = t + lane * 16;
    float v[16];
    #pragma unroll
    for (int q = 0; q < 4; ++q) {
      float4 c = *(const float4*)(crow + base + q * 4);
      float4 a = *(const float4*)(sv + base + q * 4);
      v[q * 4 + 0] = fmaf(c.x, -SCL_F, a.x);
      v[q * 4 + 1] = fmaf(c.y, -SCL_F, a.y);
      v[q * 4 + 2] = fmaf(c.z, -SCL_F, a.z);
      v[q * 4 + 3] = fmaf(c.w, -SCL_F, a.w);
    }
    float cm = v[0];
    #pragma unroll
    for (int q = 1; q < 16; ++q) cm = fmaxf(cm, v[q]);
    if (cm > m) { s *= fast_exp2(m - cm); m = cm; }   // rare rescale
    #pragma unroll
    for (int q = 0; q < 16; ++q) s += fast_exp2(v[q] - m);
  }

  #pragma unroll
  for (int off = 16; off; off >>= 1) {
    float mo = __shfl_xor(m, off, 32);
    float so = __shfl_xor(s, off, 32);
    float M  = fmaxf(m, mo);
    s = s * fast_exp2(m - M) + so * fast_exp2(mo - M);
    m = M;
  }
  if (lane == 0)
    vout[row] = -EPS_F * (logw + LN2_F * (m + fast_log2(s)));
}

// ------------------------------------------------------------------
// helpers: zero, (sum f + sum g)/N, final relu combine
// ------------------------------------------------------------------
__global__ void zero_kernel(float* __restrict__ p, int n) {
  int i = blockIdx.x * blockDim.x + threadIdx.x;
  if (i < n) p[i] = 0.0f;
}

__global__ void sum_fg_kernel(const float* __restrict__ fg,
                              float* __restrict__ scal, int idx) {
  __shared__ float red[256];
  float s = 0.0f;
  for (int i = threadIdx.x; i < 2 * NROWS; i += 256) s += fg[i];
  red[threadIdx.x] = s;
  __syncthreads();
  for (int w = 128; w; w >>= 1) {
    if ((int)threadIdx.x < w) red[threadIdx.x] += red[threadIdx.x + w];
    __syncthreads();
  }
  if (threadIdx.x == 0) scal[idx] = red[0] * (1.0f / (float)NROWS);
}

__global__ void final_combine_kernel(const float* __restrict__ scal,
                                     float* __restrict__ out) {
  out[0] = fmaxf(scal[0] - 0.5f * scal[1] - 0.5f * scal[2], 0.0f);
}

// ------------------------------------------------------------------
// launcher
// ------------------------------------------------------------------
extern "C" void kernel_launch(void* const* d_in, const int* in_sizes, int n_in,
                              void* d_out, int out_size, void* d_ws, size_t ws_size,
                              hipStream_t stream) {
  const float* x = (const float*)d_in[0];
  const float* y = (const float*)d_in[1];
  float* out = (float*)d_out;
  char* ws = (char*)d_ws;

  // workspace layout
  float*    bufA = (float*)(ws);                              // 64 MB  (C)
  float*    bufB = (float*)(ws + ((size_t)64 << 20));         // 64 MB  (C^T)
  _Float16* xh   = (_Float16*)(ws + ((size_t)128 << 20));     // 4 MB
  _Float16* yh   = xh + (size_t)NROWS * KDIM;                 // 4 MB
  float*    x2   = (float*)(yh + (size_t)NROWS * KDIM);       // 16 KB
  float*    y2   = x2 + NROWS;                                // 16 KB
  float*    f    = y2 + NROWS;                                // 16 KB
  float*    g    = f + NROWS;                                 // 16 KB (adjacent to f)
  float*    scal = g + NROWS;                                 // 3 floats

  const float logw = -8.31776616672610f; // -log(4096)
  const int nel = NROWS * KDIM;

  cvt_f16_kernel<<<(nel + 255) / 256, 256, 0, stream>>>(x, xh, nel);
  cvt_f16_kernel<<<(nel + 255) / 256, 256, 0, stream>>>(y, yh, nel);
  row_norms_kernel<<<NROWS / 8, 256, 0, stream>>>(x, x2);
  row_norms_kernel<<<NROWS / 8, 256, 0, stream>>>(y, y2);

  dim3 ggrid(NROWS / 64, NROWS / 128);

  // ---- OT(x, y): need C and C^T (two GEMMs); both stay L2-resident ----
  gemm_cost_kernel<<<ggrid, 256, 0, stream>>>(xh, yh, x2, y2, bufA);
  gemm_cost_kernel<<<ggrid, 256, 0, stream>>>(yh, xh, y2, x2, bufB);
  zero_kernel<<<(2 * NROWS + 255) / 256, 256, 0, stream>>>(f, 2 * NROWS);
  for (int it = 0; it < 50; ++it) {
    lse_rows_kernel<<<NROWS / 8, 256, 0, stream>>>(bufA, g, f, logw);
    lse_rows_kernel<<<NROWS / 8, 256, 0, stream>>>(bufB, f, g, logw);
  }
  sum_fg_kernel<<<1, 256, 0, stream>>>(f, scal, 0);

  // ---- OT(x, x): symmetric cost, C == C^T ----
  gemm_cost_kernel<<<ggrid, 256, 0, stream>>>(xh, xh, x2, x2, bufA);
  zero_kernel<<<(2 * NROWS + 255) / 256, 256, 0, stream>>>(f, 2 * NROWS);
  for (int it = 0; it < 50; ++it) {
    lse_rows_kernel<<<NROWS / 8, 256, 0, stream>>>(bufA, g, f, logw);
    lse_rows_kernel<<<NROWS / 8, 256, 0, stream>>>(bufA, f, g, logw);
  }
  sum_fg_kernel<<<1, 256, 0, stream>>>(f, scal, 1);

  // ---- OT(y, y): symmetric cost ----
  gemm_cost_kernel<<<ggrid, 256, 0, stream>>>(yh, yh, y2, y2, bufA);
  zero_kernel<<<(2 * NROWS + 255) / 256, 256, 0, stream>>>(f, 2 * NROWS);
  for (int it = 0; it < 50; ++it) {
    lse_rows_kernel<<<NROWS / 8, 256, 0, stream>>>(bufA, g, f, logw);
    lse_rows_kernel<<<NROWS / 8, 256, 0, stream>>>(bufA, f, g, logw);
  }
  sum_fg_kernel<<<1, 256, 0, stream>>>(f, scal, 2);

  final_combine_kernel<<<1, 1, 0, stream>>>(scal, out);
}
